// InitSynthesisBlock_84172769067143
// MI455X (gfx1250) — matrix operand order repaired and checked by reading, not compiled, not run on hardware
//
#include <hip/hip_runtime.h>
#include <hip/hip_bf16.h>

// ---------------------------------------------------------------------------
// StyleGAN init synthesis block, fused for MI455X (gfx1250, wave32, WMMA).
// B=4096, LATENT=512, C=512, S=4. Conv (309 GFLOP) dominates -> bf16 WMMA
// (v_wmma_f32_16x16x32_bf16) implicit GEMM with f32 accumulate; everything
// else fused around it.
// ---------------------------------------------------------------------------

typedef __attribute__((ext_vector_type(16))) __bf16 v16bf;
typedef __attribute__((ext_vector_type(8)))  float  v8f;

union Frag {            // 32 bytes = one WMMA 16-bit operand per lane
    v16bf v;
    uint4 q[2];
};

#define BATCH   4096
#define LATENT  512
#define CCH     512
#define SP      16          // 4x4 spatial
#define KSTEPS  144         // (9 taps * 512 ci) / 32
#define XROWSTR 520         // LDS row stride in halves (1040B -> 4-bank skew)
#define EPS     1e-5f

__device__ __forceinline__ float red16(float x) {
    // reduce across a 16-lane half of the wave (xor masks < 16 stay in-half)
    x += __shfl_xor(x, 8, 32);
    x += __shfl_xor(x, 4, 32);
    x += __shfl_xor(x, 2, 32);
    x += __shfl_xor(x, 1, 32);
    return x;
}

__device__ __forceinline__ unsigned short f2bf(float f) {
    union { __bf16 h; unsigned short u; } cv;
    cv.h = (__bf16)f;
    return cv.u;
}

// --- prep: f32 -> bf16 bit-copy ---------------------------------------------
__global__ void cvt_bf16_kernel(const float* __restrict__ src,
                                unsigned short* __restrict__ dst, int n) {
    int i = blockIdx.x * 256 + threadIdx.x;
    if (i < n) dst[i] = f2bf(src[i]);
}

// --- prep: pack conv_w [co][ci][3][3] into per-lane WMMA A-fragment layout --
// packedA[ct][ks][lane][e], e -> K = ks*32 + e + lgrp*8 + (e>=8 ? 8 : 0),
// K = tap*512 + ci,  row co = ct*16 + (lane&15).
__global__ void pack_conv_kernel(const float* __restrict__ cw,
                                 unsigned short* __restrict__ pa) {
    int idx = blockIdx.x * 256 + threadIdx.x;
    if (idx >= 32 * KSTEPS * 32 * 16) return;
    int e    = idx & 15;
    int lane = (idx >> 4) & 31;
    int rest = idx >> 9;
    int ks   = rest % KSTEPS;
    int ct   = rest / KSTEPS;
    int lgrp = lane >> 4;
    int co   = ct * 16 + (lane & 15);
    int K    = ks * 32 + e + lgrp * 8 + (e >= 8 ? 8 : 0);
    int tap  = K >> 9;
    int ci   = K & 511;
    pa[idx] = f2bf(cw[(co * CCH + ci) * 9 + tap]);
}

// --- styles = w @ style_w^T + style_b, one wave per 16x16 tile --------------
__global__ __launch_bounds__(256)
void styles_kernel(const unsigned short* __restrict__ wbf,
                   const unsigned short* __restrict__ sw1,
                   const unsigned short* __restrict__ sw2,
                   const float* __restrict__ sb1,
                   const float* __restrict__ sb2,
                   float* __restrict__ st1, float* __restrict__ st2) {
    const int lane = threadIdx.x & 31;
    const int wv   = threadIdx.x >> 5;
    const int ln   = lane & 15;
    const int lgrp = lane >> 4;

    int tile  = blockIdx.x * 8 + wv;          // 0 .. 32767
    int setId = tile >> 14;                   // / (64*256)
    int rem   = tile & 16383;
    int jt    = rem >> 8;                     // style-row tile (64)
    int bt    = rem & 255;                    // batch tile (256)

    const unsigned short* sw = setId ? sw2 : sw1;
    const float*          sb = setId ? sb2 : sb1;
    float*                st = setId ? st2 : st1;

    const int m = jt * 16 + ln;               // style output row j
    const int n = bt * 16 + ln;               // batch column b

    v8f acc = {};
#pragma unroll 4
    for (int ks = 0; ks < LATENT / 32; ++ks) {
        int kb = ks * 32;
        Frag af, bf;
        af.q[0] = *(const uint4*)&sw[m * LATENT + kb + lgrp * 8];
        af.q[1] = *(const uint4*)&sw[m * LATENT + kb + 16 + lgrp * 8];
        const uint4* pb = (const uint4*)&wbf[n * LATENT + kb + lgrp * 8];
        bf.q[0] = pb[0];
        bf.q[1] = pb[2];
        acc = __builtin_amdgcn_wmma_f32_16x16x32_bf16(
            false, af.v, false, bf.v, (short)0, acc, false, false);
    }
#pragma unroll
    for (int r = 0; r < 8; ++r) {
        int j = jt * 16 + r + lgrp * 8;       // C/D: M = r + 8*lanegrp
        int b = bt * 16 + ln;                 // C/D: N = lane&15
        st[b * (2 * CCH) + j] = acc[r] + sb[j];
    }
}

// --- fused: build x1 (bias+noise+leaky+IN+AdaIN1) -> implicit-GEMM conv ->
//            bias+noise2+leaky+IN+AdaIN2 -> out.  2 samples / workgroup. ----
__global__ __launch_bounds__(256)
void fused_conv_kernel(const float* __restrict__ noise1,
                       const float* __restrict__ noise2,
                       const float* __restrict__ init_c,
                       const float* __restrict__ bias_init,
                       const float* __restrict__ nw1,
                       const float* __restrict__ nw2,
                       const float* __restrict__ conv_b,
                       const unsigned short* __restrict__ packedA,
                       const float* __restrict__ styles1,
                       const float* __restrict__ styles2,
                       float* __restrict__ out) {
    __shared__ __attribute__((aligned(16))) unsigned short ldsX[2 * SP * XROWSTR];

    const int tid  = threadIdx.x;
    const int lane = tid & 31;
    const int wv   = tid >> 5;                // 8 waves
    const int ln   = lane & 15;
    const int lgrp = lane >> 4;
    const int b0   = blockIdx.x * 2;

    // ---- Phase 0: normalized AdaIN-1 activations -> LDS (bf16, [pos][ci])
    for (int task = tid; task < 2 * CCH; task += 256) {
        int smp = task >> 9;
        int ci  = task & (CCH - 1);
        int b   = b0 + smp;
        unsigned short* xT = &ldsX[smp * SP * XROWSTR];

        float bias = bias_init[ci];
        float nw   = nw1[ci];
        float v[SP];
        float mu = 0.f;
#pragma unroll
        for (int p = 0; p < SP; ++p) {
            float x = init_c[ci * SP + p] + bias + nw * noise1[b * SP + p];
            x = (x >= 0.f) ? x : 0.2f * x;    // leaky_relu(0.2)
            v[p] = x;
            mu += x;
        }
        mu *= (1.f / 16.f);
        float var = 0.f;
#pragma unroll
        for (int p = 0; p < SP; ++p) { float d = v[p] - mu; var += d * d; }
        var *= (1.f / 16.f);
        float inv = rsqrtf(var + EPS);
        float sA = styles1[b * (2 * CCH) + ci];
        float sB = styles1[b * (2 * CCH) + CCH + ci];
        float scl = (1.f + sA) * inv;
#pragma unroll
        for (int p = 0; p < SP; ++p)
            xT[p * XROWSTR + ci] = f2bf((v[p] - mu) * scl + sB);
    }
    __syncthreads();

    // ---- Phase 1: implicit GEMM, M=co (4 tiles/wave), N=16 positions (2 smp)
    v8f acc[2][4];
#pragma unroll
    for (int s = 0; s < 2; ++s)
#pragma unroll
        for (int c = 0; c < 4; ++c) acc[s][c] = (v8f){};

    const int oh = ln >> 2, ow = ln & 3;      // this lane's output position

#pragma unroll 4
    for (int ks = 0; ks < KSTEPS; ++ks) {
        int tap    = ks >> 4;                 // 16 k-steps per tap
        int cibase = (ks & 15) * 32;
        int dh = tap / 3 - 1, dw = tap % 3 - 1;
        int ih = oh + dh, iw = ow + dw;
        bool valid = (ih >= 0) & (ih < 4) & (iw >= 0) & (iw < 4);
        int q = (valid ? (ih * 4 + iw) : 0);
        int coff = cibase + lgrp * 8;

        Frag bf[2];
#pragma unroll
        for (int smp = 0; smp < 2; ++smp) {
            const uint4* src =
                (const uint4*)&ldsX[smp * SP * XROWSTR + q * XROWSTR + coff];
            uint4 q0 = src[0], q1 = src[2];
            if (!valid) { q0 = (uint4){0,0,0,0}; q1 = (uint4){0,0,0,0}; }
            bf[smp].q[0] = q0;
            bf[smp].q[1] = q1;
        }
#pragma unroll
        for (int c = 0; c < 4; ++c) {
            int ct = wv * 4 + c;
            const unsigned short* abase =
                &packedA[(((ct * KSTEPS + ks) * 32) + lane) * 16];
            // prefetch this co-tile's A fragment 8 k-steps ahead
            // (32 lanes x 32B blanket the 1KB tile -> global_prefetch_b8)
            if (ks + 8 < KSTEPS)
                __builtin_prefetch(abase + 8 * 32 * 16, 0, 1);
            const uint4* asrc = (const uint4*)abase;
            Frag af;
            af.q[0] = asrc[0];
            af.q[1] = asrc[1];
#pragma unroll
            for (int smp = 0; smp < 2; ++smp)
                acc[smp][c] = __builtin_amdgcn_wmma_f32_16x16x32_bf16(
                    false, af.v, false, bf[smp].v, (short)0,
                    acc[smp][c], false, false);
        }
    }

    // ---- Phase 2: epilogue in registers; IN-2 via half-wave shuffle reduce
#pragma unroll
    for (int smp = 0; smp < 2; ++smp) {
        int b = b0 + smp;
        float n2 = noise2[b * SP + ln];
#pragma unroll
        for (int c = 0; c < 4; ++c) {
            int ct = wv * 4 + c;
#pragma unroll
            for (int r = 0; r < 8; ++r) {
                int co = ct * 16 + r + lgrp * 8;
                float x = acc[smp][c][r] + conv_b[co] + nw2[co] * n2;
                x = (x >= 0.f) ? x : 0.2f * x;
                float mu  = red16(x) * 0.0625f;
                float d   = x - mu;
                float var = red16(d * d) * 0.0625f;
                float inv = rsqrtf(var + EPS);
                float sA = styles2[b * (2 * CCH) + co];
                float sB = styles2[b * (2 * CCH) + CCH + co];
                out[(b * CCH + co) * SP + ln] = d * inv * (1.f + sA) + sB;
            }
        }
    }
}

// ---------------------------------------------------------------------------
extern "C" void kernel_launch(void* const* d_in, const int* in_sizes, int n_in,
                              void* d_out, int out_size, void* d_ws, size_t ws_size,
                              hipStream_t stream) {
    const float* w      = (const float*)d_in[0];
    const float* noise1 = (const float*)d_in[1];
    const float* noise2 = (const float*)d_in[2];
    const float* init_c = (const float*)d_in[3];
    const float* bias_i = (const float*)d_in[4];
    const float* nw1    = (const float*)d_in[5];
    const float* nw2    = (const float*)d_in[6];
    const float* sw1    = (const float*)d_in[7];
    const float* sb1    = (const float*)d_in[8];
    const float* cw     = (const float*)d_in[9];
    const float* cb     = (const float*)d_in[10];
    const float* sw2    = (const float*)d_in[11];
    const float* sb2    = (const float*)d_in[12];

    // workspace layout (all 16B aligned): ~44.6 MB total
    unsigned short* w_bf    = (unsigned short*)d_ws;                 // 4096*512
    unsigned short* sw1_bf  = w_bf   + BATCH * LATENT;               // 1024*512
    unsigned short* sw2_bf  = sw1_bf + 2 * CCH * LATENT;             // 1024*512
    unsigned short* packedA = sw2_bf + 2 * CCH * LATENT;             // 32*144*32*16
    float* st1 = (float*)(packedA + 32 * KSTEPS * 32 * 16);          // 4096*1024
    float* st2 = st1 + BATCH * 2 * CCH;                              // 4096*1024

    cvt_bf16_kernel<<<(BATCH * LATENT + 255) / 256, 256, 0, stream>>>(
        w, w_bf, BATCH * LATENT);
    cvt_bf16_kernel<<<(2 * CCH * LATENT + 255) / 256, 256, 0, stream>>>(
        sw1, sw1_bf, 2 * CCH * LATENT);
    cvt_bf16_kernel<<<(2 * CCH * LATENT + 255) / 256, 256, 0, stream>>>(
        sw2, sw2_bf, 2 * CCH * LATENT);
    pack_conv_kernel<<<(32 * KSTEPS * 32 * 16 + 255) / 256, 256, 0, stream>>>(
        cw, packedA);
    styles_kernel<<<4096, 256, 0, stream>>>(
        w_bf, sw1_bf, sw2_bf, sb1, sb2, st1, st2);
    fused_conv_kernel<<<BATCH / 2, 256, 0, stream>>>(
        noise1, noise2, init_c, bias_i, nw1, nw2, cb,
        packedA, st1, st2, (float*)d_out);
}
